// MacGNN_89953795047803
// MI455X (gfx1250) — matrix operand order repaired and checked by reading, not compiled, not run on hardware
//
#include <hip/hip_runtime.h>
#include <hip/hip_bf16.h>
#include <stdint.h>

// ---------------------------------------------------------------------------
// MacGNN forward for MI455X (gfx1250, wave32, WMMA).
// neigh_agg collapses to (query @ Vw) because its softmax is over a single key.
// All GEMMs use v_wmma_f32_16x16x32_f16 (f16 operands, f32 accumulate).
// Global weights are pre-packed into B-fragment-major layout so each B operand
// is two global_load_b128 instead of 16 strided u16 loads.
// ---------------------------------------------------------------------------

typedef __attribute__((ext_vector_type(16))) _Float16 v16h;
typedef __attribute__((ext_vector_type(8)))  float    v8f;

union HV { v16h v; _Float16 h[16]; };

#define XCOLS 472
#define TAU   0.8f

// ---- f16 packed-weight arena layout (offsets in halfs, inside d_ws) ----
// all matrices stored fragment-major: frag f = kt*(N/16)+ct, 512 halfs each
#define VW_I      0                 // i_agg.Vw   K=32,N=64
#define VW_U      2048              // u_agg.Vw
#define VW_CB     4096              // cb_agg.Vw
#define ATT_USER  6144              // Wq,Wk,Wv,Wo each K=64,N=64
#define ATT_ITEM  22528
#define WIHT_I    38912             // Wih^T K=64,N=192
#define WIHT_U    51200
#define W0H       63488             // K=576,N=256
#define W1H       210944            // K=256,N=256
#define W2H       276480            // K=256,N=128
#define W3H       309248            // K=128,N=80
#define WH_HALFS  319488
#define HBUF_OFF  ((size_t)WH_HALFS * 2)   // bytes; h concat buffer (B x 576 f32)

// ---- dynamic LDS layout for the fused sequence kernel (bytes) ----
#define OFF_EMB   0        // f16 [128*32]
#define OFF_WROW  8192     // f32 [128]
#define OFF_XH    8704     // f16 [128*64]
#define OFF_QH    25088    // f16 [128*64]
#define OFF_KH    41472    // f16 [128*64]
#define OFF_VH    57856    // f16 [128*64]
#define OFF_SH    74240    // f16 [128*128]
#define OFF_OH    107008   // f16 [128*64]
#define OFF_GH    123392   // f16 [128*64]
#define OFF_GX    139776   // f16 [128*192]
#define OFF_GHS   188928   // f32 [192]
#define OFF_HV    189696   // f32 [64]
#define SEQ_LDS   189952

#define MLP_LDS   (64*576*2 + 64*256*2)   // A0 (64x576 f16) + A1 (64x256 f16)

// ---------------------------------------------------------------------------
// WMMA fragment helpers (gfx1250 wave32; A/B in this code's contiguous k order, C/D as in ISA 7.12.2)
// ---------------------------------------------------------------------------
// A (16x32 f16): lane l<16 -> M=l, K = k0+{0..15};
//                lane l>=16 -> M=l-16, K = k0+{16..31}
static __device__ inline v16h ld_a32(const _Float16* s, int lda, int row0, int k0, int lane) {
    int r  = row0 + (lane & 15);
    int kb = k0 + ((lane >> 4) << 4);
    HV a;
#pragma unroll
    for (int j = 0; j < 8; ++j) {
        a.h[j]     = s[r * lda + kb + j];
        a.h[8 + j] = s[r * lda + kb + 8 + j];
    }
    return a.v;
}

// A with logical K=16 (zero-padded to 32): used for per-head Q (hd=16)
static __device__ inline v16h ld_a16(const _Float16* s, int lda, int row0, int k0, int lane) {
    int r  = row0 + (lane & 15);
    int kb = k0;
    HV a;
#pragma unroll
    for (int j = 0; j < 8; ++j) {
        a.h[j]     = (lane < 16) ? s[r * lda + kb + j] : (_Float16)0.f;
        a.h[8 + j] = (lane < 16) ? s[r * lda + kb + 8 + j] : (_Float16)0.f;
    }
    return a.v;
}

// Generic B (32x16 f16) from LDS: b.h[j] = s[(k0 + (l>>4)*16 + j)*ldb + col0 + l%16]
static __device__ inline v16h ld_b32(const _Float16* s, int ldb, int k0, int col0, int lane) {
    int c  = col0 + (lane & 15);
    int kb = k0 + ((lane >> 4) << 4);
    HV b;
#pragma unroll
    for (int j = 0; j < 16; ++j) b.h[j] = s[(kb + j) * ldb + c];
    return b.v;
}

// Packed B fragment from global: one aligned v16h (2x global_load_b128) per lane
static __device__ inline v16h ld_b32p(const _Float16* Wp, int ntiles, int kt, int ct, int lane) {
    const v16h* p = (const v16h*)(Wp + (((kt * ntiles + ct) << 5) + lane) * 16);
    return *p;
}

// B = Kmat^T for attention scores: B[kk][n] = s[(col0+n)*ld + d0 + kk], K=16 padded
static __device__ inline v16h ld_bT16(const _Float16* s, int ld, int col0, int d0, int lane) {
    HV b;
    if (lane < 16) {
        int c = col0 + lane;
#pragma unroll
        for (int j = 0; j < 16; ++j) b.h[j] = s[c * ld + d0 + j];
    } else {
#pragma unroll
        for (int j = 0; j < 16; ++j) b.h[j] = (_Float16)0.f;
    }
    return b.v;
}

// C/D (16x16 f32): lane l -> N=l%16, VGPR r -> M = r + 8*(l/16)
static __device__ inline void st_c_f16(_Float16* d, int ldd, int row0, int col0,
                                       const v8f& c, int lane) {
    int r0 = row0 + ((lane >> 4) << 3);
    int cc = col0 + (lane & 15);
#pragma unroll
    for (int r = 0; r < 8; ++r) d[(r0 + r) * ldd + cc] = (_Float16)c[r];
}

static __device__ inline v8f wmma32(v16h a, v16h b, v8f c) {
    return __builtin_amdgcn_wmma_f32_16x16x32_f16(false, a, false, b, (short)0, c, false, false);
}

// ---------------------------------------------------------------------------
// Weight prep: pack f32 (K x N row-major, or (N x K) if transposed) into
// B-fragment-major f16: dst[f*512 + lane*16 + j] with
//   k = kt*32 + (lane>>4)*16 + j ,  n = ct*16 + (lane&15) ,  f = kt*(N/16)+ct
// ---------------------------------------------------------------------------
__global__ void k_packB(const float* __restrict__ src, _Float16* __restrict__ dst,
                        int K, int N, int transposed) {
    int i = blockIdx.x * 256 + threadIdx.x;
    if (i >= K * N) return;
    int f = i >> 9, r = i & 511;
    int lane = r >> 4, j = r & 15;
    int ntiles = N >> 4;
    int kt = f / ntiles, ct = f % ntiles;
    int k = kt * 32 + ((lane >> 4) << 4) + j;
    int n = ct * 16 + (lane & 15);
    float v = transposed ? src[n * K + k] : src[k * N + n];
    dst[i] = (_Float16)v;
}

// copy user/item embeddings into the concat buffer
__global__ void k_emb(const int* __restrict__ x, const float* __restrict__ ue,
                      const float* __restrict__ ie, float* __restrict__ hbuf) {
    int b = blockIdx.x, t = threadIdx.x;
    const int* xr = x + (long)b * XCOLS;
    if (t < 32) {
        hbuf[(long)b * 576 + t]       = ue[(long)xr[0]   * 32 + t];
        hbuf[(long)b * 576 + 224 + t] = ie[(long)xr[231] * 32 + t];
    }
}

// ---------------------------------------------------------------------------
// Fused per-(batch,sequence) kernel:
//   gather -> X = emb@Vw * weight -> MHA -> tanh -> conv-sum | GRU -> h column
// grid = (8, B), block = 256 (8 waves)
// ---------------------------------------------------------------------------
__global__ void __launch_bounds__(256)
k_seq(const int* __restrict__ x,
      const float* __restrict__ user_rel, const float* __restrict__ item_rel,
      const float* __restrict__ u_macro,  const float* __restrict__ i_macro,
      const _Float16* __restrict__ wh,
      const float* __restrict__ whh_i, const float* __restrict__ whh_u,
      const float* __restrict__ w_user_conv, const float* __restrict__ w_item_conv,
      const float* __restrict__ w_cbi_conv,  const float* __restrict__ w_cbu_conv,
      float* __restrict__ hbuf)
{
    extern __shared__ char smem[];
    const int seq  = blockIdx.x;
    const int b    = blockIdx.y;
    const int tid  = threadIdx.x;
    const int lane = tid & 31;
    const int wv   = tid >> 5;

    // zero the whole LDS arena (gives us all zero-padding for free)
    {
        uint32_t* z = (uint32_t*)smem;
        for (int i = tid; i < SEQ_LDS / 4; i += 256) z[i] = 0u;
    }
    __syncthreads();

    // ---- per-sequence configuration ----
    int L = 0, wmode = 2, cnt0 = 0, idx0 = 0, post = 0, outcol = 0;
    const float *macro = nullptr, *gtab = nullptr, *convw = nullptr, *whh = nullptr;
    const _Float16 *vw = wh, *attw = wh, *wihT = wh;
    switch (seq) {
    case 0: L = 65;  macro = i_macro; wmode = 0; cnt0 = 1;   vw = wh + VW_I;  attw = wh + ATT_ITEM; post = 0; convw = w_item_conv; outcol = 32;  break;
    case 1: L = 65;  macro = u_macro; wmode = 0; cnt0 = 66;  vw = wh + VW_U;  attw = wh + ATT_USER; post = 0; convw = w_user_conv; outcol = 96;  break;
    case 2: L = 65;  macro = u_macro; wmode = 0; cnt0 = 232; vw = wh + VW_U;  attw = wh + ATT_USER; post = 0; convw = w_user_conv; outcol = 256; break;
    case 3: L = 65;  macro = i_macro; wmode = 0; cnt0 = 297; vw = wh + VW_I;  attw = wh + ATT_ITEM; post = 0; convw = w_item_conv; outcol = 320; break;
    case 4: L = 100; gtab = item_rel; wmode = 1; idx0 = 131; vw = wh + VW_I;  attw = wh + ATT_ITEM; post = 1; wihT = wh + WIHT_I; whh = whh_i; outcol = 160; break;
    case 5: L = 110; gtab = user_rel; wmode = 1; idx0 = 362; vw = wh + VW_U;  attw = wh + ATT_USER; post = 1; wihT = wh + WIHT_U; whh = whh_u; outcol = 384; break;
    case 6: L = 10;  gtab = item_rel; wmode = 2; idx0 = 462; vw = wh + VW_I;  attw = wh + ATT_ITEM; post = 0; convw = w_cbi_conv;  outcol = 448; break;
    default:L = 10;  gtab = item_rel; wmode = 2; idx0 = 462; vw = wh + VW_CB; attw = wh + ATT_ITEM; post = 0; convw = w_cbu_conv;  outcol = 512; break;
    }
    const int RT = (L + 15) >> 4;
    const int* xr = x + (long)b * XCOLS;

    _Float16* embD = (_Float16*)(smem + OFF_EMB);
    float*    wrow = (float*)   (smem + OFF_WROW);
    _Float16* Xh   = (_Float16*)(smem + OFF_XH);
    _Float16* qh   = (_Float16*)(smem + OFF_QH);
    _Float16* kh   = (_Float16*)(smem + OFF_KH);
    _Float16* vh   = (_Float16*)(smem + OFF_VH);
    _Float16* Sh   = (_Float16*)(smem + OFF_SH);
    _Float16* oh   = (_Float16*)(smem + OFF_OH);
    _Float16* Gh   = (_Float16*)(smem + OFF_GH);
    _Float16* gx   = (_Float16*)(smem + OFF_GX);
    float*    ghs  = (float*)   (smem + OFF_GHS);
    float*    hv   = (float*)   (smem + OFF_HV);

    // ---- gather embeddings (f32 -> f16 in LDS) ----
    for (int i = tid; i < L * 32; i += 256) {
        int l = i >> 5, d = i & 31;
        float v;
        if (macro) v = macro[i];
        else       v = gtab[(long)xr[idx0 + l] * 32 + d];
        embD[l * 32 + d] = (_Float16)v;
    }
    // ---- per-row weight (macro softmax / positivity mask / 1) ----
    if (wmode == 0) {
        if (tid < L) wrow[tid] = __logf((float)xr[cnt0 + tid] + 1.0f) * (1.0f / TAU);
        __syncthreads();
        if (tid == 0) {
            float m = -1e30f;
            for (int l = 0; l < L; ++l) m = fmaxf(m, wrow[l]);
            float s = 0.f;
            for (int l = 0; l < L; ++l) { float e = __expf(wrow[l] - m); wrow[l] = e; s += e; }
            float inv = 1.0f / s;
            for (int l = 0; l < L; ++l) wrow[l] *= inv;
        }
    } else if (wmode == 1) {
        if (tid < L) wrow[tid] = (xr[idx0 + tid] > 0) ? 1.0f : 0.0f;
    } else {
        if (tid < L) wrow[tid] = 1.0f;
    }
    __syncthreads();

    // ---- X = (emb @ Vw) * wrow   (L x 64, K = 32) ----
    for (int t = wv; t < RT * 4; t += 8) {
        int rt = t >> 2, ct = t & 3;
        v8f c = {};
        c = wmma32(ld_a32(embD, 32, rt * 16, 0, lane), ld_b32p(vw, 4, 0, ct, lane), c);
        int r0 = rt * 16 + ((lane >> 4) << 3), cc = ct * 16 + (lane & 15);
#pragma unroll
        for (int r = 0; r < 8; ++r)
            Xh[(r0 + r) * 64 + cc] = (_Float16)(c[r] * wrow[r0 + r]);
    }
    __syncthreads();

    // ---- Q,K,V = X @ {Wq,Wk,Wv}   (K = 64) ----
    for (int t = wv; t < RT * 4 * 3; t += 8) {
        int m = t / (RT * 4), tt = t % (RT * 4);
        int rt = tt >> 2, ct = tt & 3;
        const _Float16* W = attw + m * 4096;
        _Float16* dst = (m == 0) ? qh : (m == 1) ? kh : vh;
        v8f c = {};
#pragma unroll
        for (int ks = 0; ks < 2; ++ks)
            c = wmma32(ld_a32(Xh, 64, rt * 16, ks * 32, lane),
                       ld_b32p(W, 4, ks, ct, lane), c);
        st_c_f16(dst, 64, rt * 16, ct * 16, c, lane);
    }
    __syncthreads();

    // ---- per-head attention ----
    for (int hh = 0; hh < 4; ++hh) {
        // scores = q_h @ k_h^T / sqrt(16)
        for (int t = wv; t < RT * RT; t += 8) {
            int rt = t / RT, ct = t % RT;
            v8f c = {};
            c = wmma32(ld_a16(qh, 64, rt * 16, hh * 16, lane),
                       ld_bT16(kh, 64, ct * 16, hh * 16, lane), c);
            int r0 = rt * 16 + ((lane >> 4) << 3), cc = ct * 16 + (lane & 15);
#pragma unroll
            for (int r = 0; r < 8; ++r)
                Sh[(r0 + r) * 128 + cc] = (_Float16)(c[r] * 0.25f);
        }
        __syncthreads();
        // row softmax over valid columns (padding columns stay exactly 0)
        if (tid < L) {
            _Float16* srow = Sh + tid * 128;
            float m = -1e30f;
            for (int l = 0; l < L; ++l) m = fmaxf(m, (float)srow[l]);
            float s = 0.f;
            for (int l = 0; l < L; ++l) s += __expf((float)srow[l] - m);
            float inv = 1.0f / s;
            for (int l = 0; l < L; ++l)
                srow[l] = (_Float16)(__expf((float)srow[l] - m) * inv);
        }
        __syncthreads();
        // o_h = S @ v_h   (K padded to 128; pad region is zero)
        for (int t = wv; t < RT; t += 8) {
            v8f c = {};
#pragma unroll
            for (int ks = 0; ks < 4; ++ks)
                c = wmma32(ld_a32(Sh, 128, t * 16, ks * 32, lane),
                           ld_b32(vh, 64, ks * 32, hh * 16, lane), c);
            st_c_f16(oh, 64, t * 16, hh * 16, c, lane);
        }
        __syncthreads();
    }

    // ---- out = tanh(o @ Wo) ----
    for (int t = wv; t < RT * 4; t += 8) {
        int rt = t >> 2, ct = t & 3;
        v8f c = {};
#pragma unroll
        for (int ks = 0; ks < 2; ++ks)
            c = wmma32(ld_a32(oh, 64, rt * 16, ks * 32, lane),
                       ld_b32p(attw + 3 * 4096, 4, ks, ct, lane), c);
        int r0 = rt * 16 + ((lane >> 4) << 3), cc = ct * 16 + (lane & 15);
#pragma unroll
        for (int r = 0; r < 8; ++r)
            Gh[(r0 + r) * 64 + cc] = (_Float16)tanhf(c[r]);
    }
    __syncthreads();

    if (post == 0) {
        // conv-sum: out_e = sum_l Gh[l,e] * convw[l]
        if (tid < 64) {
            float s = 0.f;
            for (int l = 0; l < L; ++l) s += (float)Gh[l * 64 + tid] * convw[l];
            hbuf[(long)b * 576 + outcol + tid] = s;
        }
    } else {
        // gx = Gh @ Wih^T   (L x 192, K = 64)
        for (int t = wv; t < RT * 12; t += 8) {
            int rt = t / 12, ct = t % 12;
            v8f c = {};
#pragma unroll
            for (int ks = 0; ks < 2; ++ks)
                c = wmma32(ld_a32(Gh, 64, rt * 16, ks * 32, lane),
                           ld_b32p(wihT, 12, ks, ct, lane), c);
            st_c_f16(gx, 192, rt * 16, ct * 16, c, lane);
        }
        __syncthreads();
        // GRU scan
        if (tid < 64) hv[tid] = 0.f;
        __syncthreads();
        for (int t = 0; t < L; ++t) {
            if (tid < 192) {
                const float* wr = whh + tid * 64;
                float s = 0.f;
#pragma unroll 4
                for (int k2 = 0; k2 < 64; ++k2) s += wr[k2] * hv[k2];
                ghs[tid] = s;
            }
            __syncthreads();
            if (tid < 64) {
                float g0 = (float)gx[t * 192 + tid];
                float g1 = (float)gx[t * 192 + 64 + tid];
                float g2 = (float)gx[t * 192 + 128 + tid];
                float r  = 1.f / (1.f + __expf(-(g0 + ghs[tid])));
                float zz = 1.f / (1.f + __expf(-(g1 + ghs[64 + tid])));
                float n  = tanhf(g2 + r * ghs[128 + tid]);
                hv[tid] = (1.f - zz) * n + zz * hv[tid];
            }
            __syncthreads();
        }
        if (tid < 64) hbuf[(long)b * 576 + outcol + tid] = hv[tid];
    }
}

// ---------------------------------------------------------------------------
// MLP kernel: 64 batch rows per block, all layers resident in LDS
// ---------------------------------------------------------------------------
static __device__ void mlp_layer(const _Float16* A, int lda, const _Float16* W,
                                 const float* bias, _Float16* C, int ldc, int K, int N,
                                 int lane, int wv) {
    int cts = N >> 4, nt = 4 * cts;
    for (int t = wv; t < nt; t += 8) {
        int rt = t / cts, ct = t % cts;
        v8f c = {};
        for (int ks = 0; ks < K / 32; ++ks)
            c = wmma32(ld_a32(A, lda, rt * 16, ks * 32, lane),
                       ld_b32p(W, cts, ks, ct, lane), c);
        int r0 = rt * 16 + ((lane >> 4) << 3), cc = ct * 16 + (lane & 15);
#pragma unroll
        for (int r = 0; r < 8; ++r) {
            float v = c[r] + bias[cc];
            C[(r0 + r) * ldc + cc] = (_Float16)fmaxf(v, 0.f);
        }
    }
}

__global__ void __launch_bounds__(256)
k_mlp(const float* __restrict__ hbuf, const _Float16* __restrict__ wh,
      const float* __restrict__ b0, const float* __restrict__ b1,
      const float* __restrict__ b2, const float* __restrict__ b3,
      const float* __restrict__ W4, const float* __restrict__ b4,
      float* __restrict__ out)
{
    extern __shared__ char smem[];
    _Float16* A0 = (_Float16*)smem;                       // 64 x 576
    _Float16* A1 = (_Float16*)(smem + 64 * 576 * 2);      // 64 x 256
    int tid = threadIdx.x, lane = tid & 31, wv = tid >> 5;
    long row0 = (long)blockIdx.x * 64;

    for (int i = tid; i < 64 * 576; i += 256)
        A0[i] = (_Float16)hbuf[row0 * 576 + i];
    __syncthreads();

    mlp_layer(A0, 576, wh + W0H, b0, A1, 256, 576, 256, lane, wv); __syncthreads();
    mlp_layer(A1, 256, wh + W1H, b1, A0, 256, 256, 256, lane, wv); __syncthreads();
    mlp_layer(A0, 256, wh + W2H, b2, A1, 128, 256, 128, lane, wv); __syncthreads();
    mlp_layer(A1, 128, wh + W3H, b3, A0,  80, 128,  80, lane, wv); __syncthreads();

    if (tid < 64) {
        float s = b4[0];
        for (int k = 0; k < 80; ++k) s += (float)A0[tid * 80 + k] * W4[k];
        out[row0 + tid] = 1.f / (1.f + __expf(-s));
    }
}

// ---------------------------------------------------------------------------
// Host launcher
// Input order (setup_inputs flattening): x, then params in insertion order.
// ---------------------------------------------------------------------------
extern "C" void kernel_launch(void* const* d_in, const int* in_sizes, int n_in,
                              void* d_out, int out_size, void* d_ws, size_t ws_size,
                              hipStream_t stream) {
    (void)in_sizes; (void)n_in; (void)out_size; (void)ws_size;

    const int*   x         = (const int*)  d_in[0];
    const float* user_emb  = (const float*)d_in[1];
    const float* item_emb  = (const float*)d_in[2];
    const float* user_rel  = (const float*)d_in[3];
    const float* item_rel  = (const float*)d_in[4];
    const float* u_macro   = (const float*)d_in[5];
    const float* i_macro   = (const float*)d_in[6];
    const float* u_agg_Vw  = (const float*)d_in[9];
    const float* i_agg_Vw  = (const float*)d_in[12];
    const float* cb_agg_Vw = (const float*)d_in[15];
    // user_attn Wq,Wk,Wv,Wo = d_in[16..19]; item_attn = d_in[20..23]
    const float* w_user_conv = (const float*)d_in[24];
    const float* w_item_conv = (const float*)d_in[25];
    const float* w_cbi_conv  = (const float*)d_in[26];
    const float* w_cbu_conv  = (const float*)d_in[27];
    const float* wih_i = (const float*)d_in[28];
    const float* whh_i = (const float*)d_in[29];
    const float* wih_u = (const float*)d_in[30];
    const float* whh_u = (const float*)d_in[31];
    const float* W0 = (const float*)d_in[32]; const float* b0 = (const float*)d_in[33];
    const float* W1 = (const float*)d_in[34]; const float* b1 = (const float*)d_in[35];
    const float* W2 = (const float*)d_in[36]; const float* b2 = (const float*)d_in[37];
    const float* W3 = (const float*)d_in[38]; const float* b3 = (const float*)d_in[39];
    const float* W4 = (const float*)d_in[40]; const float* b4 = (const float*)d_in[41];

    _Float16* wh   = (_Float16*)d_ws;
    float*    hbuf = (float*)((char*)d_ws + HBUF_OFF);
    float*    out  = (float*)d_out;

    const int B = 2048;
    auto blocks = [](int n) { return (n + 255) / 256; };

    // --- pack all GEMM weights into B-fragment-major f16 ---
    k_packB<<<blocks(2048), 256, 0, stream>>>(i_agg_Vw,  wh + VW_I,  32, 64, 0);
    k_packB<<<blocks(2048), 256, 0, stream>>>(u_agg_Vw,  wh + VW_U,  32, 64, 0);
    k_packB<<<blocks(2048), 256, 0, stream>>>(cb_agg_Vw, wh + VW_CB, 32, 64, 0);
    for (int j = 0; j < 4; ++j) {
        k_packB<<<blocks(4096), 256, 0, stream>>>((const float*)d_in[16 + j], wh + ATT_USER + j * 4096, 64, 64, 0);
        k_packB<<<blocks(4096), 256, 0, stream>>>((const float*)d_in[20 + j], wh + ATT_ITEM + j * 4096, 64, 64, 0);
    }
    k_packB<<<blocks(64 * 192), 256, 0, stream>>>(wih_i, wh + WIHT_I, 64, 192, 1);
    k_packB<<<blocks(64 * 192), 256, 0, stream>>>(wih_u, wh + WIHT_U, 64, 192, 1);
    k_packB<<<blocks(576 * 256), 256, 0, stream>>>(W0, wh + W0H, 576, 256, 0);
    k_packB<<<blocks(256 * 256), 256, 0, stream>>>(W1, wh + W1H, 256, 256, 0);
    k_packB<<<blocks(256 * 128), 256, 0, stream>>>(W2, wh + W2H, 256, 128, 0);
    k_packB<<<blocks(128 * 80),  256, 0, stream>>>(W3, wh + W3H, 128,  80, 0);

    // --- embedding columns of the concat buffer ---
    k_emb<<<B, 64, 0, stream>>>(x, user_emb, item_emb, hbuf);

    // --- fused gather/aggregate/MHA/conv-GRU, one workgroup per (seq, batch) ---
    k_seq<<<dim3(8, B), 256, SEQ_LDS, stream>>>(
        x, user_rel, item_rel, u_macro, i_macro, wh, whh_i, whh_u,
        w_user_conv, w_item_conv, w_cbi_conv, w_cbu_conv, hbuf);

    // --- MLP head ---
    k_mlp<<<B / 64, 256, MLP_LDS, stream>>>(hbuf, wh, b0, b1, b2, b3, W4, b4, out);
}
